// LiftedConv_23527830848172
// MI455X (gfx1250) — matrix operand order repaired and checked by reading, not compiled, not run on hardware
//
#include <hip/hip_runtime.h>

typedef __attribute__((ext_vector_type(16))) _Float16 v16h;
typedef __attribute__((ext_vector_type(8)))  _Float16 v8h;
typedef __attribute__((ext_vector_type(8)))  float    v8f;
typedef __attribute__((ext_vector_type(4)))  int      v4i;

#define H_IMG  256
#define W_IMG  256
#define NBATCH 8
#define ICT    32          // C_IN * N_ROT
#define OCT    64          // N_ROT * C_OUT
#define KS     5
#define NSPAT  25
#define TILE_W 128
#define LCOLS  (TILE_W + 4)   // 132
#define LPITCH 40             // padded channel pitch: 80B row stride, 16B aligned chunks

#if __has_builtin(__builtin_amdgcn_global_load_async_to_lds_b128) && \
    __has_builtin(__builtin_amdgcn_s_wait_asynccnt)
#define USE_ASYNC_LDS 1
#else
#define USE_ASYNC_LDS 0
#endif

__device__ __forceinline__ float sgnf(float x) {
    return (x > 0.f) ? 1.f : ((x < 0.f) ? -1.f : 0.f);
}

// quadratic B-spline exactly as in the reference
__device__ __forceinline__ float B2f(float x) {
    float a = -3.f * (x - 0.5f) * (x - 0.5f) * sgnf(0.5f - x);
    float b =        (x - 1.5f) * (x - 1.5f) * sgnf(1.5f - x);
    float c = -3.f * (1.f + 2.f * x) * (1.f + 2.f * x) * sgnf(0.5f + x) * 0.25f;
    float d =        (3.f + 2.f * x) * (3.f + 2.f * x) * sgnf(1.5f + x) * 0.25f;
    return (a + b + c + d) * 0.25f;
}

// 32-byte A-fragment slice copy: async global->LDS if available, else VGPR bounce
__device__ __forceinline__ void stage_a_slice(const _Float16* gsrc, _Float16* ldst) {
#if USE_ASYNC_LDS
    typedef __attribute__((address_space(1))) v4i gv4i;   // global int4*
    typedef __attribute__((address_space(3))) v4i lv4i;   // LDS int4*
    __builtin_amdgcn_global_load_async_to_lds_b128((gv4i*)gsrc,       (lv4i*)ldst,       0, 0);
    __builtin_amdgcn_global_load_async_to_lds_b128((gv4i*)(gsrc + 8), (lv4i*)(ldst + 8), 0, 0);
#else
    *(v8h*)ldst       = *(const v8h*)gsrc;
    *(v8h*)(ldst + 8) = *(const v8h*)(gsrc + 8);
#endif
}

__device__ __forceinline__ void wait_async_copies() {
#if USE_ASYNC_LDS
    __builtin_amdgcn_s_wait_asynccnt(0);
#endif
}

// ---------------------------------------------------------------------------
// Kernel 1: build the lifted filter bank directly in WMMA A-fragment layout.
// A is M=64 (oc) x K=800, K = spatial_tap*32 + channel.  Stored f16 as
// Af[((t*25 + s)*32 + lane)*16 + j]  (t = oc tile, s = spatial tap).
// ---------------------------------------------------------------------------
__global__ __launch_bounds__(256) void build_afrag_kernel(
    const float* __restrict__ Wt,     // (4, 200, 8)
    _Float16*    __restrict__ Af) {
    const int gid = blockIdx.x * 256 + threadIdx.x;   // 0 .. 51199
    const int oc  = gid / 800;
    const int k   = gid % 800;
    const int s   = k >> 5;            // spatial tap 0..24
    const int ic  = k & 31;            // channel 0..31
    const int kh  = s / 5, kw = s % 5;
    const int thi = oc >> 3;           // rotation index of vmapped theta
    const int o   = oc & 7;            // small output channel
    const int ci  = ic >> 3;           // C_IN index
    const int r   = ic & 7;            // input rotation index

    const float PI  = 3.14159265358979323846f;
    const float QP  = PI * 0.25f;      // 2*pi/8
    const float theta = -QP * (float)thi;

    float sn, cs;
    __sincosf(theta, &sn, &cs);

    const float gx = (float)(kh - 2), gy = (float)(kw - 2);
    const float xs = gx * cs - gy * sn;
    const float ys = gx * sn + gy * cs;

    // rotation spline values: B2(angle_dist(r*QP + theta, q*QP) / QP)
    float rv[8];
    const float ra = (float)r * QP + theta;
    #pragma unroll
    for (int q = 0; q < 8; ++q) {
        float d = (float)q * QP - ra + PI;
        d -= floorf(d * (1.f / (2.f * PI))) * (2.f * PI);   // python-style mod
        d -= PI;
        rv[q] = B2f(d / QP);
    }

    float acc = 0.f;
    for (int p = 0; p < 25; ++p) {
        const float cx = (float)(p / 5 - 2);
        const float cy = (float)(p % 5 - 2);
        const float xv = B2f(cx - xs) * B2f(cy - ys);
        const float* wp = Wt + (ci * 200 + p * 8) * 8 + o;
        float ss = 0.f;
        #pragma unroll
        for (int q = 0; q < 8; ++q) ss += rv[q] * wp[q * 8];
        acc += xv * ss;
    }

    // scatter into 16x32 f16 A-fragment lane order (ISA 7.12.2 table)
    const int t    = oc >> 4;
    const int row  = oc & 15;
    const int lane = row + (((ic >> 3) & 1) << 4);
    const int j    = (ic & 7) | ((ic & 16) >> 1);
    Af[((t * NSPAT + s) * 32 + lane) * 16 + j] = (_Float16)acc;
}

// ---------------------------------------------------------------------------
// Kernel 2: implicit-GEMM 5x5 conv via v_wmma_f32_16x16x32_f16.
// Block = 128 threads (4 waves), tile = 128 pixels of one output row.
// A K-slices double-buffered in LDS via async global->LDS copies.
// Wave: 32 pixels x 64 oc = 8 accumulators; 200 WMMAs per wave.
// ---------------------------------------------------------------------------
__global__ __launch_bounds__(128) void lifted_conv_wmma_kernel(
    const float*    __restrict__ X,    // (8, 32, 256, 256)
    const _Float16* __restrict__ Af,   // fragment-ordered filter bank
    float*          __restrict__ out)  // (8, 8, 8, 256, 256)
{
    __shared__ __attribute__((aligned(16))) _Float16 lds[KS * LCOLS * LPITCH];
    __shared__ __attribute__((aligned(16))) _Float16 abuf[2][4 * 32 * 16];  // 2 x 4KB

    const int tid = threadIdx.x;
    const int w0  = blockIdx.x * TILE_W;
    const int h   = blockIdx.y;
    const int b   = blockIdx.z;

    const int wave = tid >> 5;
    const int lane = tid & 31;
    const int n    = lane & 15;    // B/D column within tile
    const int kseg = lane >> 4;    // K half selector
    const int px0  = wave * 32;
    const int aslc = (wave * 32 + lane) * 16;   // this thread's 32B slice in abuf

    // ---- stage input patch (h-2..h+2, w0-2..w0+129, 32ch) as f16 into LDS ----
    // layout: lds[(row*LCOLS + col)*LPITCH + ic]; loop order keeps global
    // reads coalesced along w (col fastest).
    for (int e = tid; e < KS * LCOLS * ICT; e += 128) {   // 21120 = 165 * 128
        const int col = e % LCOLS;
        const int t2  = e / LCOLS;
        const int ic  = t2 & 31;
        const int row = t2 >> 5;
        const int gh  = h + row - 2;
        const int gw  = w0 + col - 2;
        float v = 0.f;
        if ((unsigned)gh < H_IMG && (unsigned)gw < W_IMG)
            v = X[((b * ICT + ic) * H_IMG + gh) * W_IMG + gw];
        lds[(row * LCOLS + col) * LPITCH + ic] = (_Float16)v;
    }

    // prologue: async-stage A K-slice for s=0 (wave t copies oc-tile t)
    stage_a_slice(Af + (wave * NSPAT * 32 + lane) * 16, &abuf[0][aslc]);

    v8f acc[2][4];
    #pragma unroll
    for (int g = 0; g < 2; ++g)
        #pragma unroll
        for (int t = 0; t < 4; ++t)
            acc[g][t] = (v8f){0.f, 0.f, 0.f, 0.f, 0.f, 0.f, 0.f, 0.f};

    for (int s = 0; s < NSPAT; ++s) {
        wait_async_copies();       // this wave's async LDS writes have landed
        __syncthreads();           // everyone's writes (and X staging) visible

        if (s + 1 < NSPAT) {       // overlap next A slice copy with compute
            stage_a_slice(Af + ((wave * NSPAT + s + 1) * 32 + lane) * 16,
                          &abuf[(s + 1) & 1][aslc]);
            __builtin_prefetch(Af + ((s + 2 < NSPAT ? s + 2 : s + 1) * 32 + lane) * 16, 0, 3);
        }

        const _Float16* ab = abuf[s & 1];
        const int kh = s / 5, kw = s % 5;

        // B fragments: K x N = 32ch x 16px; lane holds ch kseg*8..+7 and 16+kseg*8..+7
        v16h bfr[2];
        #pragma unroll
        for (int g = 0; g < 2; ++g) {
            const v8h* p = (const v8h*)&lds[(kh * LCOLS + px0 + g * 16 + n + kw) * LPITCH];
            v16h bb;
            ((v8h*)&bb)[0] = p[kseg];
            ((v8h*)&bb)[1] = p[2 + kseg];
            bfr[g] = bb;
        }

        #pragma unroll
        for (int t = 0; t < 4; ++t) {
            const v8h* ap = (const v8h*)&ab[(t * 32 + lane) * 16];
            v16h a;
            ((v8h*)&a)[0] = ap[0];
            ((v8h*)&a)[1] = ap[1];
            acc[0][t] = __builtin_amdgcn_wmma_f32_16x16x32_f16(
                false, a, false, bfr[0], (short)0, acc[0][t], false, false);
            acc[1][t] = __builtin_amdgcn_wmma_f32_16x16x32_f16(
                false, a, false, bfr[1], (short)0, acc[1][t], false, false);
        }
    }

    // ---- epilogue: scale by 2*pi/8, permute (r,co) -> (co,r), store ----
    const float scale = 0.78539816339744831f;
    #pragma unroll
    for (int g = 0; g < 2; ++g) {
        const int w = w0 + px0 + g * 16 + n;
        #pragma unroll
        for (int t = 0; t < 4; ++t) {
            const int rrot = t * 2 + kseg;           // oc>>3 with oc = t*16+kseg*8+v
            #pragma unroll
            for (int v = 0; v < 8; ++v) {            // co = v
                const int chan = (b * 8 + v) * 8 + rrot;
                out[chan * (H_IMG * W_IMG) + h * W_IMG + w] = scale * acc[g][t][v];
            }
        }
    }
}

extern "C" void kernel_launch(void* const* d_in, const int* in_sizes, int n_in,
                              void* d_out, int out_size, void* d_ws, size_t ws_size,
                              hipStream_t stream) {
    const float* X  = (const float*)d_in[0];   // (8, 4, 8, 256, 256) f32
    const float* Wt = (const float*)d_in[1];   // (4, 200, 8) f32
    float*   out = (float*)d_out;              // (8, 8, 8, 256, 256) f32
    _Float16* Af = (_Float16*)d_ws;            // 51200 f16 = 100 KB scratch

    build_afrag_kernel<<<dim3(200), dim3(256), 0, stream>>>(Wt, Af);
    lifted_conv_wmma_kernel<<<dim3(W_IMG / TILE_W, H_IMG, NBATCH),
                              dim3(128), 0, stream>>>(X, Af, out);
}
